// TurboQuantLinear_19550691131800
// MI455X (gfx1250) — matrix operand order repaired and checked by reading, not compiled, not run on hardware
//
#include <hip/hip_runtime.h>

// ---- clang ext_vector types (trivial, union/bit_cast safe) ----
typedef __attribute__((ext_vector_type(16))) __bf16        v16bf;
typedef __attribute__((ext_vector_type(8)))  __bf16        v8bf;
typedef __attribute__((ext_vector_type(4)))  __bf16        v4bf;
typedef __attribute__((ext_vector_type(8)))  float         v8f;
typedef __attribute__((ext_vector_type(4)))  float         v4f;
typedef __attribute__((ext_vector_type(4)))  int           v4i;
typedef __attribute__((ext_vector_type(4)))  unsigned int  v4u;
typedef __attribute__((ext_vector_type(2)))  unsigned int  v2u;

#define M_DIM 8192
#define N_DIM 11008
#define K_DIM 4096
#define N_GROUPS 32

// fused-fallback tile
#define BM 128
#define BN 128
#define BK 64
#define LDS_K (BK + 8)

// main GEMM tile
#define GBM 128
#define GBN 256
#define GBK 64
#define GLK (GBK + 8)    // 144B rows, 16B aligned

union FragCast { v4u u[2]; v16bf v; };

// ---- CDNA5 async global->LDS copy (ASYNCcnt path), guarded ----
#if __has_builtin(__builtin_amdgcn_global_load_async_to_lds_b128)
#define HAVE_ASYNC 1
#else
#define HAVE_ASYNC 0
#endif

#if HAVE_ASYNC
typedef __attribute__((address_space(1))) v4i* as1_v4i_p;
typedef __attribute__((address_space(3))) v4i* as3_v4i_p;
#endif

__device__ __forceinline__ void async_copy16(const unsigned short* g, unsigned short* l) {
#if HAVE_ASYNC
    __builtin_amdgcn_global_load_async_to_lds_b128(
        (as1_v4i_p)(v4i*)const_cast<unsigned short*>(g),
        (as3_v4i_p)(v4i*)l,
        0, 0);
#else
    *(v4u*)l = *(const v4u*)g;   // sync fallback: compiler manages waits
#endif
}

__device__ __forceinline__ void wait_async() {
#if HAVE_ASYNC
#if __has_builtin(__builtin_amdgcn_s_wait_asynccnt)
    __builtin_amdgcn_s_wait_asynccnt(0);
#else
    asm volatile("s_wait_asynccnt 0x0" ::: "memory");
#endif
#endif
}

// ============================================================================
// Pass 1: x (f32) -> bf16
// ============================================================================
__global__ __launch_bounds__(256)
void convert_x_bf16(const float* __restrict__ x, unsigned short* __restrict__ xb)
{
    size_t idx = (size_t)blockIdx.x * 256 + threadIdx.x;
    const v4f* xv = (const v4f*)x;
    v4f a = xv[idx * 2 + 0];
    v4f b = xv[idx * 2 + 1];
    v4bf ha = __builtin_convertvector(a, v4bf);
    v4bf hb = __builtin_convertvector(b, v4bf);
    v2u pa = __builtin_bit_cast(v2u, ha);
    v2u pb = __builtin_bit_cast(v2u, hb);
    v4u p; p.x = pa.x; p.y = pa.y; p.z = pb.x; p.w = pb.y;
    *(v4u*)(xb + idx * 8) = p;
}

// ============================================================================
// Pass 2: qweight int4 -> bf16 W
// ============================================================================
__global__ __launch_bounds__(256)
void dequant_w_bf16(const int*   __restrict__ qweight,
                    const float* __restrict__ scales,
                    const int*   __restrict__ qzeros,
                    unsigned short* __restrict__ wb)
{
    int t = blockIdx.x * 256 + threadIdx.x;
    int o = t >> 7;
    int c = (t & 127) * 4;
    int g = c >> 4;
    float s = scales[o * N_GROUPS + g];
    float z = (float)qzeros[o * N_GROUPS + g];
    float nzs = -z * s;

    v4i q = *(const v4i*)&qweight[(size_t)o * (K_DIM / 8) + c];
    int qa[4] = { q.x, q.y, q.z, q.w };
    unsigned short* dst = wb + (size_t)o * K_DIM + c * 8;
#pragma unroll
    for (int j = 0; j < 4; ++j) {
        v8f f;
#pragma unroll
        for (int i = 0; i < 8; ++i)
            f[i] = fmaf((float)((qa[j] >> (4 * i)) & 15), s, nzs);
        v8bf h = __builtin_convertvector(f, v8bf);
        *(v4u*)(dst + j * 8) = __builtin_bit_cast(v4u, h);
    }
}

// ============================================================================
// Pass 3: bf16 WMMA GEMM, 128x256 block / 64x64 per wave, async double-buffer
// ============================================================================
__global__ __launch_bounds__(256, 2)
void bf16_wmma_gemm_async(const unsigned short* __restrict__ xb,
                          const unsigned short* __restrict__ wb,
                          float* __restrict__ out)
{
    __shared__ unsigned short As[2][GBM][GLK];
    __shared__ unsigned short Bs[2][GBN][GLK];

    const int tid    = threadIdx.x;
    const int lane   = tid & 31;
    const int wave   = tid >> 5;
    const int wave_m = wave >> 2;   // 0..1 -> 64 rows
    const int wave_n = wave & 3;    // 0..3 -> 64 cols

    const int m_base = blockIdx.y * GBM;
    const int n_base = blockIdx.x * GBN;

    const int half = lane >> 4;
    const int ml   = lane & 15;

    // copy mapping: 16B chunks; A tile = 1024 chunks (4/thread), B tile = 2048 (8/thread)
    const int lrow = tid >> 3;
    const int lkc  = (tid & 7) * 8;

    v8f acc[4][4];
#pragma unroll
    for (int mi = 0; mi < 4; ++mi)
#pragma unroll
        for (int ni = 0; ni < 4; ++ni)
            acc[mi][ni] = v8f{};

    auto issue_stage = [&](int k0, int buf) {
#pragma unroll
        for (int i = 0; i < 4; ++i) {
            int row = lrow + i * 32;
            async_copy16(&xb[(size_t)(m_base + row) * K_DIM + k0 + lkc],
                         &As[buf][row][lkc]);
        }
#pragma unroll
        for (int i = 0; i < 8; ++i) {
            int row = lrow + i * 32;
            async_copy16(&wb[(size_t)(n_base + row) * K_DIM + k0 + lkc],
                         &Bs[buf][row][lkc]);
        }
    };

    auto compute = [&](int buf) {
#pragma unroll
        for (int kk = 0; kk < 2; ++kk) {
            FragCast a[4];
#pragma unroll
            for (int mi = 0; mi < 4; ++mi) {
                int row = wave_m * 64 + mi * 16 + ml;
                a[mi].u[0] = *(const v4u*)&As[buf][row][kk * 32 + half * 8];
                a[mi].u[1] = *(const v4u*)&As[buf][row][kk * 32 + 16 + half * 8];
            }
            FragCast b[4];
#pragma unroll
            for (int ni = 0; ni < 4; ++ni) {
                int col = wave_n * 64 + ni * 16 + ml;
                b[ni].u[0] = *(const v4u*)&Bs[buf][col][kk * 32 + half * 8];
                b[ni].u[1] = *(const v4u*)&Bs[buf][col][kk * 32 + 16 + half * 8];
            }
#pragma unroll
            for (int mi = 0; mi < 4; ++mi)
#pragma unroll
                for (int ni = 0; ni < 4; ++ni)
                    acc[mi][ni] = __builtin_amdgcn_wmma_f32_16x16x32_bf16(
                        false, a[mi].v, false, b[ni].v,
                        (short)0, acc[mi][ni], false, false);
        }
    };

    const int STAGES = K_DIM / GBK;   // 64

    issue_stage(0, 0);
    wait_async();
    __syncthreads();

    for (int s = 0; s < STAGES; ++s) {
        if (s + 1 < STAGES) issue_stage((s + 1) * GBK, (s + 1) & 1);
        compute(s & 1);
        wait_async();        // next stage landed (this wave); barrier covers the rest
        __syncthreads();
    }

    // C/D layout: lanes 0-15 -> M=r, lanes 16-31 -> M=8+r; N = lane&15
#pragma unroll
    for (int mi = 0; mi < 4; ++mi)
#pragma unroll
        for (int ni = 0; ni < 4; ++ni) {
            int col = n_base + wave_n * 64 + ni * 16 + ml;
#pragma unroll
            for (int r = 0; r < 8; ++r) {
                int row = m_base + wave_m * 64 + mi * 16 + half * 8 + r;
                out[(size_t)row * N_DIM + col] = acc[mi][ni][r];
            }
        }
}

// ============================================================================
// Fallback: fused dequant + GEMM (if workspace too small)
// ============================================================================
__device__ __forceinline__ unsigned short f2bf(float f) {
    unsigned u = __builtin_bit_cast(unsigned, f);
    u += 0x7FFFu + ((u >> 16) & 1u);
    return (unsigned short)(u >> 16);
}

__device__ __forceinline__ v4u dequant8(int q, float s, float z) {
    unsigned short h[8];
#pragma unroll
    for (int i = 0; i < 8; ++i)
        h[i] = f2bf(((float)((q >> (4 * i)) & 15) - z) * s);
    v4u r;
    r.x = (unsigned)h[0] | ((unsigned)h[1] << 16);
    r.y = (unsigned)h[2] | ((unsigned)h[3] << 16);
    r.z = (unsigned)h[4] | ((unsigned)h[5] << 16);
    r.w = (unsigned)h[6] | ((unsigned)h[7] << 16);
    return r;
}

__global__ __launch_bounds__(256, 2)
void gptq_wmma_gemm_fused(const float* __restrict__ x,
                          const int*   __restrict__ qweight,
                          const float* __restrict__ scales,
                          const int*   __restrict__ qzeros,
                          float*       __restrict__ out)
{
    __shared__ unsigned short As[BM][LDS_K];
    __shared__ unsigned short Bs[BN][LDS_K];

    const int tid  = threadIdx.x;
    const int lane = tid & 31;
    const int wave = tid >> 5;
    const int wave_m = wave >> 2;
    const int wave_n = wave & 3;
    const int m_base = blockIdx.y * BM;
    const int n_base = blockIdx.x * BN;
    const int half = lane >> 4;
    const int ml   = lane & 15;
    const int brow = tid >> 1;
    const int bqi  = (tid & 1) * 4;
    const int orow = n_base + brow;

    v8f acc[4][2];
#pragma unroll
    for (int mi = 0; mi < 4; ++mi)
#pragma unroll
        for (int ni = 0; ni < 2; ++ni)
            acc[mi][ni] = v8f{};

    for (int k0 = 0; k0 < K_DIM; k0 += BK) {
        const v4f* xv = (const v4f*)x;
#pragma unroll
        for (int i = 0; i < 8; ++i) {
            int j = tid + i * 256;
            int row = j >> 4;
            int kq  = j & 15;
            v4f v = xv[(size_t)(m_base + row) * (K_DIM / 4) + (k0 >> 2) + kq];
            v4bf h = __builtin_convertvector(v, v4bf);
            *(v2u*)&As[row][kq * 4] = __builtin_bit_cast(v2u, h);
        }
        {
            const int g = k0 >> 7;
            const float s = scales[orow * N_GROUPS + g];
            const float z = (float)qzeros[orow * N_GROUPS + g];
            v4i q4 = *(const v4i*)&qweight[(size_t)orow * (K_DIM / 8) + (k0 >> 3) + bqi];
            *(v4u*)&Bs[brow][(bqi + 0) * 8] = dequant8(q4.x, s, z);
            *(v4u*)&Bs[brow][(bqi + 1) * 8] = dequant8(q4.y, s, z);
            *(v4u*)&Bs[brow][(bqi + 2) * 8] = dequant8(q4.z, s, z);
            *(v4u*)&Bs[brow][(bqi + 3) * 8] = dequant8(q4.w, s, z);
        }
        __syncthreads();
#pragma unroll
        for (int kk = 0; kk < 2; ++kk) {
            FragCast a[4];
#pragma unroll
            for (int mi = 0; mi < 4; ++mi) {
                int row = wave_m * 64 + mi * 16 + ml;
                a[mi].u[0] = *(const v4u*)&As[row][kk * 32 + half * 8];
                a[mi].u[1] = *(const v4u*)&As[row][kk * 32 + 16 + half * 8];
            }
            FragCast b[2];
#pragma unroll
            for (int ni = 0; ni < 2; ++ni) {
                int col = wave_n * 32 + ni * 16 + ml;
                b[ni].u[0] = *(const v4u*)&Bs[col][kk * 32 + half * 8];
                b[ni].u[1] = *(const v4u*)&Bs[col][kk * 32 + 16 + half * 8];
            }
#pragma unroll
            for (int mi = 0; mi < 4; ++mi)
#pragma unroll
                for (int ni = 0; ni < 2; ++ni)
                    acc[mi][ni] = __builtin_amdgcn_wmma_f32_16x16x32_bf16(
                        false, a[mi].v, false, b[ni].v,
                        (short)0, acc[mi][ni], false, false);
        }
        __syncthreads();
    }

#pragma unroll
    for (int mi = 0; mi < 4; ++mi)
#pragma unroll
        for (int ni = 0; ni < 2; ++ni) {
            int col = n_base + wave_n * 32 + ni * 16 + ml;
#pragma unroll
            for (int r = 0; r < 8; ++r) {
                int row = m_base + wave_m * 64 + mi * 16 + half * 8 + r;
                out[(size_t)row * N_DIM + col] = acc[mi][ni][r];
            }
        }
}

// ============================================================================
extern "C" void kernel_launch(void* const* d_in, const int* in_sizes, int n_in,
                              void* d_out, int out_size, void* d_ws, size_t ws_size,
                              hipStream_t stream) {
    const float* x       = (const float*)d_in[0];
    const int*   qweight = (const int*)d_in[1];
    const float* scales  = (const float*)d_in[2];
    const int*   qzeros  = (const int*)d_in[3];
    float* out = (float*)d_out;

    const size_t xb_elems = (size_t)M_DIM * K_DIM;
    const size_t wb_elems = (size_t)N_DIM * K_DIM;
    const size_t need = (xb_elems + wb_elems) * sizeof(unsigned short); // 150 MiB

    if (ws_size >= need) {
        unsigned short* xb = (unsigned short*)d_ws;
        unsigned short* wb = xb + xb_elems;
        convert_x_bf16<<<(unsigned)(xb_elems / 8 / 256), 256, 0, stream>>>(x, xb);
        dequant_w_bf16<<<(unsigned)(N_DIM * (K_DIM / 8 / 4) / 256), 256, 0, stream>>>(
            qweight, scales, qzeros, wb);
        dim3 grid(N_DIM / GBN, M_DIM / GBM);   // (43, 64)
        bf16_wmma_gemm_async<<<grid, 256, 0, stream>>>(xb, wb, out);
    } else {
        dim3 grid(N_DIM / BN, M_DIM / BM);     // (86, 64)
        gptq_wmma_gemm_fused<<<grid, 256, 0, stream>>>(x, qweight, scales, qzeros, out);
    }
}
